// MambaBlock_CD_8211977470542
// MI455X (gfx1250) — matrix operand order repaired and checked by reading, not compiled, not run on hardware
//
#include <hip/hip_runtime.h>
#include <hip/hip_bf16.h>

// ---------------- problem constants ----------------
#define DMQ   256                 // model dim
#define DINQ  512                 // inner dim
#define LSEQ  4096
#define BATCH 2
#define MROWS (BATCH * LSEQ)      // 8192 GEMM rows
#define DPROJ 48                  // DTR + 2*DS
#define NCH   8                   // scan chunks per chain
#define LC    (LSEQ / NCH)        // 512 steps per chunk

typedef __attribute__((ext_vector_type(16))) __bf16 v16bf;
typedef __attribute__((ext_vector_type(8)))  float  v8f;

__device__ __forceinline__ unsigned short f2bf(float f) {
  unsigned int u = __float_as_uint(f);
  u += 0x7FFFu + ((u >> 16) & 1u);          // round-to-nearest-even
  return (unsigned short)(u >> 16);
}
__device__ __forceinline__ float silu_f(float x) { return x / (1.f + __expf(-x)); }

// ---------------- weight packing into WMMA B-fragment order ----------------
// W is (N x K) row-major f32 computing out = x @ W^T, so B[k][n] = W[n][k].
// bf16 B-fragment (32x16) in this code's contiguous k order (same as A): lanes 0-15 hold K=k0..k0+15 (elem j -> k0+j),
// lanes 16-31 hold K=k0+16..k0+31 (elem j -> k0+16+j); n = lane%16.
// Packed layout: Bp[((nt*KT + kt)*32 + lane)*16 + j] (contiguous 32B per lane),
// zero-padded out to NtPad tiles so a wave can always process 4 n-tiles.
__global__ void pack_w(const float* __restrict__ W, unsigned short* __restrict__ Bp,
                       int N, int K, int NtPad) {
  int idx = blockIdx.x * blockDim.x + threadIdx.x;
  if (idx >= NtPad * 16 * K) return;
  int j    = idx & 15;
  int lane = (idx >> 4) & 31;
  int t    = idx >> 9;               // nt*KT + kt
  int KT   = K >> 5;
  int kt   = t % KT;
  int nt   = t / KT;
  int n = nt * 16 + (lane & 15);
  int k = kt * 32 + ((lane >> 4) ? 16 + j : j);
  Bp[idx] = (n < N) ? f2bf(W[(size_t)n * K + k]) : (unsigned short)0;
}

// A = -exp(A_log), 512*16 elements
__global__ void neg_exp_k(const float* __restrict__ a, float* __restrict__ o) {
  int i = blockIdx.x * blockDim.x + threadIdx.x;
  o[i] = -__expf(a[i]);
}

// ---------------- RMSNorm -> bf16 (both streams), one wave per row ----------------
__global__ void rmsnorm_bf16(const float* __restrict__ t1, const float* __restrict__ t2,
                             const float* __restrict__ w, unsigned short* __restrict__ xn) {
  const int lane = threadIdx.x & 31;
  const size_t row = (size_t)blockIdx.x * 8 + (threadIdx.x >> 5);   // 0..16383
  const float* src = (row < MROWS) ? (t1 + row * DMQ) : (t2 + (row - MROWS) * DMQ);
  float4 v0 = *(const float4*)(src + lane * 8);
  float4 v1 = *(const float4*)(src + lane * 8 + 4);
  float ss = v0.x*v0.x + v0.y*v0.y + v0.z*v0.z + v0.w*v0.w
           + v1.x*v1.x + v1.y*v1.y + v1.z*v1.z + v1.w*v1.w;
#pragma unroll
  for (int s = 16; s; s >>= 1) ss += __shfl_xor(ss, s, 32);
  const float sc = rsqrtf(ss * (1.f / DMQ) + 1e-5f);
  const float* wp = w + lane * 8;
  union { unsigned short s[8]; uint4 q; } o;
  o.s[0] = f2bf(v0.x * sc * wp[0]); o.s[1] = f2bf(v0.y * sc * wp[1]);
  o.s[2] = f2bf(v0.z * sc * wp[2]); o.s[3] = f2bf(v0.w * sc * wp[3]);
  o.s[4] = f2bf(v1.x * sc * wp[4]); o.s[5] = f2bf(v1.y * sc * wp[5]);
  o.s[6] = f2bf(v1.z * sc * wp[6]); o.s[7] = f2bf(v1.w * sc * wp[7]);
  *(uint4*)(xn + row * DMQ + lane * 8) = o.q;
}

// ---------------- bf16 WMMA GEMM: each wave computes a 16(M) x 64(N) strip ----------------
// Per K-step: 1 A fragment (2 x b128) feeds 4 v_wmma against 4 B fragments (8 x b128):
// 10 vmem loads per 4 WMMAs, 4x A reuse vs a 16x16-per-wave scheme.
__global__ void gemm_bf16_wmma(const unsigned short* __restrict__ A,
                               const unsigned short* __restrict__ Bp,
                               const float* __restrict__ bias,
                               const float* __restrict__ resid,
                               float* __restrict__ C,
                               int Ntiles, int NtPad, int K, int ldc) {
  const int lane = threadIdx.x & 31;
  const int wave = threadIdx.x >> 5;
  const int nt0  = (blockIdx.y * 4 + wave) * 4;   // first of this wave's 4 n-tiles
  if (nt0 >= NtPad) return;                       // wave-uniform: EXEC stays all-ones
  const int mt  = blockIdx.x;
  const int KT  = K >> 5;
  const int row = lane & 15;
  const int hi  = lane >> 4;
  // A fragment: lane holds row (mt*16+row); elems 0..15 -> k0+hi*16+j
  const unsigned short* Ar = A + (size_t)(mt * 16 + row) * K + hi * 16;
  const size_t tstride = (size_t)KT * 512;        // halfs per packed n-tile
  const unsigned short* Bl = Bp + (size_t)nt0 * tstride + (size_t)lane * 16;
  v8f acc[4] = {{}, {}, {}, {}};
  for (int kt = 0; kt < KT; ++kt) {
    union { uint4 q[2]; v16bf v; } a, b0, b1, b2, b3;
    const unsigned short* ak = Ar + kt * 32;
    a.q[0] = *(const uint4*)(ak);
    a.q[1] = *(const uint4*)(ak + 8);
    const unsigned short* bk = Bl + (size_t)kt * 512;
    b0.q[0] = *(const uint4*)(bk);
    b0.q[1] = *(const uint4*)(bk + 8);
    b1.q[0] = *(const uint4*)(bk + tstride);
    b1.q[1] = *(const uint4*)(bk + tstride + 8);
    b2.q[0] = *(const uint4*)(bk + 2 * tstride);
    b2.q[1] = *(const uint4*)(bk + 2 * tstride + 8);
    b3.q[0] = *(const uint4*)(bk + 3 * tstride);
    b3.q[1] = *(const uint4*)(bk + 3 * tstride + 8);
    acc[0] = __builtin_amdgcn_wmma_f32_16x16x32_bf16(false, a.v, false, b0.v,
                                                     (short)0, acc[0], false, false);
    acc[1] = __builtin_amdgcn_wmma_f32_16x16x32_bf16(false, a.v, false, b1.v,
                                                     (short)0, acc[1], false, false);
    acc[2] = __builtin_amdgcn_wmma_f32_16x16x32_bf16(false, a.v, false, b2.v,
                                                     (short)0, acc[2], false, false);
    acc[3] = __builtin_amdgcn_wmma_f32_16x16x32_bf16(false, a.v, false, b3.v,
                                                     (short)0, acc[3], false, false);
  }
#pragma unroll
  for (int t = 0; t < 4; ++t) {
    if (nt0 + t >= Ntiles) break;                 // wave-uniform
    const int ncol = (nt0 + t) * 16 + row;
    const float badd = bias ? bias[ncol] : 0.f;
#pragma unroll
    for (int r = 0; r < 8; ++r) {
      const size_t m = (size_t)mt * 16 + r + hi * 8;  // D layout: VGPR r -> row r (+8 hi lanes)
      float v = acc[t][r] + badd;
      if (resid) v += resid[m * ldc + ncol];
      C[m * ldc + ncol] = v;
    }
  }
}

// ---------------- depthwise causal conv (DC=4) + bias + silu -> u (f32 + bf16) ----------------
__global__ void conv_silu(const float* __restrict__ xr, const float* __restrict__ cw,
                          const float* __restrict__ cb, float* __restrict__ uf,
                          unsigned short* __restrict__ ub) {
  int idx = blockIdx.x * blockDim.x + threadIdx.x;   // b*L*512 + l*512 + d
  int d = idx & 511;
  int l = (idx >> 9) & (LSEQ - 1);
  int b = idx >> 21;
  size_t mbase = (size_t)b * LSEQ * 1024;
  float acc = cb[d];
#pragma unroll
  for (int i = 0; i < 4; ++i) {
    int ll = l - 3 + i;
    if (ll >= 0) acc = fmaf(cw[d * 4 + i], xr[mbase + (size_t)ll * 1024 + d], acc);
  }
  float s = silu_f(acc);
  uf[idx] = s;
  ub[idx] = f2bf(s);
}

// ---------------- dt_proj (K=16, tiny) + softplus -> delta ----------------
__global__ void dtproj_softplus(const float* __restrict__ dbl2, const float* __restrict__ w,
                                const float* __restrict__ bias, float* __restrict__ delta) {
  int idx = blockIdx.x * blockDim.x + threadIdx.x;   // m*512 + n
  int n = idx & 511;
  size_t m = (size_t)idx >> 9;
  const float* d2 = dbl2 + m * DPROJ;                // cols 0..15 of dbl2
  float acc = bias[n];
#pragma unroll
  for (int k = 0; k < 16; ++k) acc = fmaf(d2[k], w[n * 16 + k], acc);
  delta[idx] = (acc > 20.f) ? acc : log1pf(__expf(acc));
}

// ---------------- selective scan: pass 1, per-chunk affine composition (P,Q) ----------------
// Recurrence is diagonal per (b,d,n): x = a*x + du with a = exp(delta*A[d,n]).
// Chunk composition: x_out = P*x_in + Q, P = prod(a), Q = zero-init scan.
__global__ void scan_chunk(const float* __restrict__ delta, const float* __restrict__ dbl1,
                           const float* __restrict__ dbl2, const float* __restrict__ u1,
                           const float* __restrict__ u2, const float* __restrict__ Am,
                           float* __restrict__ P, float* __restrict__ Q) {
  int gid = blockIdx.x * blockDim.x + threadIdx.x;   // 131072 threads
  int n  = gid & 15;
  int hw = gid >> 4;                                 // ((b*512+d)*8 + c)
  int c  = hw & 7;
  int d  = (hw >> 3) & 511;
  int b  = hw >> 12;
  const float An = Am[d * 16 + n];
  float p = 1.f, q = 0.f;
  size_t mbase = (size_t)b * LSEQ + c * LC;
  for (int l = 0; l < LC; ++l) {
    size_t m = mbase + l;
    float dv  = delta[m * 512 + d];
    float u1v = u1[m * 512 + d];
    float u2v = u2[m * 512 + d];
    float Bn  = dbl1[m * DPROJ + 16 + n];
    float Bpn = dbl2[m * DPROJ + 16 + n];
    float a  = __expf(dv * An);
    float du = fabsf(dv * (u1v * Bn - u2v * Bpn));
    q = fmaf(a, q, du);
    p *= a;
  }
  P[(size_t)hw * 16 + n] = p;
  Q[(size_t)hw * 16 + n] = q;
}

// ---------------- pass 2: serial prefix over the 8 chunks (per b,d,n) ----------------
__global__ void scan_prefix(const float* __restrict__ P, const float* __restrict__ Q,
                            float* __restrict__ X0) {
  int idx = blockIdx.x * blockDim.x + threadIdx.x;   // 16384 threads
  int n  = idx & 15;
  int bd = idx >> 4;
  float s = 0.f;
#pragma unroll
  for (int c = 0; c < NCH; ++c) {
    size_t o = ((size_t)bd * NCH + c) * 16 + n;
    X0[o] = s;
    s = fmaf(P[o], s, Q[o]);
  }
}

// ---------------- pass 3: replay with correct init, emit y1/y2 (with D + silu(res)) ----------------
__global__ void scan_emit(const float* __restrict__ delta, const float* __restrict__ dbl1,
                          const float* __restrict__ dbl2, const float* __restrict__ u1,
                          const float* __restrict__ u2, const float* __restrict__ Am,
                          const float* __restrict__ X0, const float* __restrict__ Dv,
                          const float* __restrict__ Dp, const float* __restrict__ xr1,
                          const float* __restrict__ xr2, unsigned short* __restrict__ y1,
                          unsigned short* __restrict__ y2) {
  int gid = blockIdx.x * blockDim.x + threadIdx.x;
  int n  = gid & 15;
  int hw = gid >> 4;
  int c  = hw & 7;
  int d  = (hw >> 3) & 511;
  int b  = hw >> 12;
  const float An  = Am[d * 16 + n];
  const float Dd  = Dv[d];
  const float Dpd = Dp[d];
  float x = X0[(size_t)hw * 16 + n];
  size_t mbase = (size_t)b * LSEQ + c * LC;
  for (int l = 0; l < LC; ++l) {
    size_t m = mbase + l;
    float dv  = delta[m * 512 + d];
    float u1v = u1[m * 512 + d];
    float u2v = u2[m * 512 + d];
    float Bn  = dbl1[m * DPROJ + 16 + n];
    float Bpn = dbl2[m * DPROJ + 16 + n];
    float Cn  = dbl1[m * DPROJ + 32 + n];
    float Cpn = dbl2[m * DPROJ + 32 + n];
    float a  = __expf(dv * An);
    float du = fabsf(dv * (u1v * Bn - u2v * Bpn));
    x = fmaf(a, x, du);
    float s1 = x * Cn;
    float s2 = x * Cpn;
#pragma unroll
    for (int s = 8; s; s >>= 1) {                 // reduce over the 16 state lanes
      s1 += __shfl_xor(s1, s, 16);
      s2 += __shfl_xor(s2, s, 16);
    }
    if (n == 0) {
      float r1 = xr1[m * 1024 + 512 + d];         // res1
      float r2 = xr2[m * 1024 + 512 + d];         // res2
      float yy1 = (s1 + u1v * Dd)  * silu_f(r1);
      float yy2 = (s2 + u2v * Dpd) * silu_f(r2);
      y1[m * 512 + d] = f2bf(yy1);
      y2[m * 512 + d] = f2bf(yy2);
    }
  }
}

// ---------------- host-side orchestration ----------------
extern "C" void kernel_launch(void* const* d_in, const int* in_sizes, int n_in,
                              void* d_out, int out_size, void* d_ws, size_t ws_size,
                              hipStream_t stream) {
  (void)in_sizes; (void)n_in; (void)out_size; (void)ws_size;
  const float* t1     = (const float*)d_in[0];
  const float* t2     = (const float*)d_in[1];
  const float* norm_w = (const float*)d_in[2];
  const float* in_w   = (const float*)d_in[3];
  const float* in_b   = (const float*)d_in[4];
  const float* conv_w = (const float*)d_in[5];
  const float* conv_b = (const float*)d_in[6];
  const float* xp_w   = (const float*)d_in[7];
  const float* dt_w   = (const float*)d_in[8];
  const float* dt_b   = (const float*)d_in[9];
  const float* A_log  = (const float*)d_in[10];
  const float* Dvp    = (const float*)d_in[11];
  const float* Dpp    = (const float*)d_in[12];
  const float* out_w  = (const float*)d_in[13];
  const float* out_b  = (const float*)d_in[14];
  float* out = (float*)d_out;

  char* wsb = (char*)d_ws;
  size_t off = 0;
  auto carve = [&](size_t bytes) -> void* {
    void* p = wsb + off;
    off += (bytes + 255) & ~(size_t)255;
    return p;
  };
  unsigned short* xn      = (unsigned short*)carve((size_t)2 * MROWS * DMQ * 2);  // 8.4 MB
  unsigned short* packIn  = (unsigned short*)carve((size_t)1024 * DMQ * 2);       // 64 tiles
  unsigned short* packXp  = (unsigned short*)carve((size_t)64 * DINQ * 2);        // 4 tiles (padded)
  unsigned short* packOut = (unsigned short*)carve((size_t)DMQ * DINQ * 2);       // 16 tiles
  float* xr1   = (float*)carve((size_t)MROWS * 1024 * 4);                         // 33.5 MB
  float* xr2   = (float*)carve((size_t)MROWS * 1024 * 4);
  float* u1f   = (float*)carve((size_t)MROWS * DINQ * 4);
  float* u2f   = (float*)carve((size_t)MROWS * DINQ * 4);
  unsigned short* u1b = (unsigned short*)carve((size_t)MROWS * DINQ * 2);
  unsigned short* u2b = (unsigned short*)carve((size_t)MROWS * DINQ * 2);
  float* dbl1  = (float*)carve((size_t)MROWS * DPROJ * 4);
  float* dbl2  = (float*)carve((size_t)MROWS * DPROJ * 4);
  float* delta = (float*)carve((size_t)MROWS * DINQ * 4);
  float* Amat  = (float*)carve((size_t)DINQ * 16 * 4);
  float* Pbuf  = (float*)carve((size_t)BATCH * DINQ * NCH * 16 * 4);
  float* Qbuf  = (float*)carve((size_t)BATCH * DINQ * NCH * 16 * 4);
  float* X0    = (float*)carve((size_t)BATCH * DINQ * NCH * 16 * 4);
  unsigned short* y1b = (unsigned short*)carve((size_t)MROWS * DINQ * 2);
  unsigned short* y2b = (unsigned short*)carve((size_t)MROWS * DINQ * 2);

  // 0) one-time prep: pack weights to B-fragment order (zero-padded), A = -exp(A_log)
  pack_w<<<1024, 256, 0, stream>>>(in_w,  packIn,  1024, DMQ,  64);   // 64*16*256
  pack_w<<<128,  256, 0, stream>>>(xp_w,  packXp,  DPROJ, DINQ, 4);   // 4*16*512
  pack_w<<<512,  256, 0, stream>>>(out_w, packOut, DMQ,   DINQ, 16);  // 16*16*512
  neg_exp_k<<<32, 256, 0, stream>>>(A_log, Amat);

  // 1) RMSNorm both streams -> bf16 activations
  rmsnorm_bf16<<<2048, 256, 0, stream>>>(t1, t2, norm_w, xn);

  // 2) in_proj GEMM (M=8192, K=256, N=1024) + bias, per stream
  gemm_bf16_wmma<<<dim3(512, 4), 128, 0, stream>>>(xn, packIn, in_b, nullptr, xr1,
                                                   64, 64, DMQ, 1024);
  gemm_bf16_wmma<<<dim3(512, 4), 128, 0, stream>>>(xn + (size_t)MROWS * DMQ, packIn, in_b,
                                                   nullptr, xr2, 64, 64, DMQ, 1024);

  // 3) depthwise causal conv + silu -> u
  conv_silu<<<16384, 256, 0, stream>>>(xr1, conv_w, conv_b, u1f, u1b);
  conv_silu<<<16384, 256, 0, stream>>>(xr2, conv_w, conv_b, u2f, u2b);

  // 4) x_proj GEMM (K=512, N=48; padded to 4 n-tiles, stores guarded)
  gemm_bf16_wmma<<<dim3(512, 1), 128, 0, stream>>>(u1b, packXp, nullptr, nullptr, dbl1,
                                                   3, 4, DINQ, DPROJ);
  gemm_bf16_wmma<<<dim3(512, 1), 128, 0, stream>>>(u2b, packXp, nullptr, nullptr, dbl2,
                                                   3, 4, DINQ, DPROJ);

  // 5) dt_proj (K=16) + softplus -> delta
  dtproj_softplus<<<16384, 256, 0, stream>>>(dbl2, dt_w, dt_b, delta);

  // 6) chunked selective scan (8 chunks): compose, prefix, replay+emit
  scan_chunk<<<512, 256, 0, stream>>>(delta, dbl1, dbl2, u1f, u2f, Amat, Pbuf, Qbuf);
  scan_prefix<<<64, 256, 0, stream>>>(Pbuf, Qbuf, X0);
  scan_emit<<<512, 256, 0, stream>>>(delta, dbl1, dbl2, u1f, u2f, Amat, X0, Dvp, Dpp,
                                     xr1, xr2, y1b, y2b);

  // 7) out_proj GEMM (K=512, N=256) + bias + residual -> d_out (out1 then out2)
  gemm_bf16_wmma<<<dim3(512, 1), 128, 0, stream>>>(y1b, packOut, out_b, t1, out,
                                                   16, 16, DINQ, DMQ);
  gemm_bf16_wmma<<<dim3(512, 1), 128, 0, stream>>>(y2b, packOut, out_b, t2,
                                                   out + (size_t)MROWS * DMQ,
                                                   16, 16, DINQ, DMQ);
}
